// Block_2430951489679
// MI455X (gfx1250) — matrix operand order repaired and checked by reading, not compiled, not run on hardware
//
#include <hip/hip_runtime.h>

typedef __bf16 bf16_t;
typedef bf16_t v16bf __attribute__((ext_vector_type(16)));
typedef float  v8f   __attribute__((ext_vector_type(8)));
typedef unsigned int v4u __attribute__((ext_vector_type(4)));
typedef int v4i_ __attribute__((ext_vector_type(4)));
typedef __attribute__((address_space(1))) v4i_* as1_v4i_ptr;   // global int4*
typedef __attribute__((address_space(3))) v4i_* as3_v4i_ptr;   // LDS int4*

union Pack16 { v4u q[2]; bf16_t e[16]; v16bf v; };

#if __has_builtin(__builtin_amdgcn_global_load_async_to_lds_b128)
#define HAVE_ASYNC_LDS 1
#else
#define HAVE_ASYNC_LDS 0
#endif

__device__ __forceinline__ bf16_t f2bf(float f) {
    unsigned u = __builtin_bit_cast(unsigned, f);
    unsigned r = u + 0x7FFFu + ((u >> 16) & 1u);
    unsigned short h = (unsigned short)(r >> 16);
    return __builtin_bit_cast(bf16_t, h);
}

// 16-byte global -> LDS copy; async (ASYNCcnt-tracked) when available.
__device__ __forceinline__ void async_cp16(const bf16_t* g, bf16_t* l) {
#if HAVE_ASYNC_LDS
    __builtin_amdgcn_global_load_async_to_lds_b128(
        (as1_v4i_ptr)g, (as3_v4i_ptr)l, 0, 0);
#else
    *(v4u*)l = *(const v4u*)g;
#endif
}

__device__ __forceinline__ void wait_async0() {
#if __has_builtin(__builtin_amdgcn_s_wait_asynccnt)
    __builtin_amdgcn_s_wait_asynccnt(0);
#else
    asm volatile("s_wait_asynccnt 0x0" ::: "memory");
#endif
}

__device__ __forceinline__ v8f wmma_bf16(v16bf a, v16bf b, v8f c) {
    // D = A(16x32 bf16) * B(32x16 bf16) + C(16x16 f32)
    return __builtin_amdgcn_wmma_f32_16x16x32_bf16(
        false, a, false, b, (short)0, c, false, false);
}

// ---------------------------------------------------------------------------
// Weight prep: W[K,N] fp32 -> WT[N,K] bf16 (transpose via LDS tile), once per
// launch. Makes every GEMM staging copy a layout-preserving 16B move.
// ---------------------------------------------------------------------------
__global__ __launch_bounds__(256)
void cvt_transpose_kernel(const float* __restrict__ W, bf16_t* __restrict__ WT,
                          int K, int N) {
    __shared__ float tile[32][33];
    const int kb = blockIdx.x * 32, nb = blockIdx.y * 32;
    const int tx = threadIdx.x & 31, ty = threadIdx.x >> 5;   // ty 0..7
    #pragma unroll
    for (int i = ty; i < 32; i += 8)
        tile[i][tx] = W[(size_t)(kb + i) * N + nb + tx];
    __syncthreads();
    #pragma unroll
    for (int i = ty; i < 32; i += 8)
        WT[(size_t)(nb + i) * K + kb + tx] = f2bf(tile[tx][i]);
}

// ---------------------------------------------------------------------------
// LayerNorm over D=768, one block (256 thr) per token row, bf16 output
// ---------------------------------------------------------------------------
__global__ __launch_bounds__(256)
void layernorm_kernel(const float* __restrict__ x, const float* __restrict__ g,
                      const float* __restrict__ beta, bf16_t* __restrict__ out) {
    const int row = blockIdx.x;
    const int tid = threadIdx.x;
    const float* xr = x + (size_t)row * 768;
    float v0 = xr[tid], v1 = xr[tid + 256], v2 = xr[tid + 512];
    __shared__ float s1[256], s2[256];
    s1[tid] = v0 + v1 + v2;
    s2[tid] = v0 * v0 + v1 * v1 + v2 * v2;
    __syncthreads();
    for (int off = 128; off; off >>= 1) {
        if (tid < off) { s1[tid] += s1[tid + off]; s2[tid] += s2[tid + off]; }
        __syncthreads();
    }
    const float mu  = s1[0] * (1.0f / 768.0f);
    const float var = s2[0] * (1.0f / 768.0f) - mu * mu;
    const float rs  = rsqrtf(var + 1e-5f);
    bf16_t* orow = out + (size_t)row * 768;
    orow[tid]       = f2bf((v0 - mu) * rs * g[tid]       + beta[tid]);
    orow[tid + 256] = f2bf((v1 - mu) * rs * g[tid + 256] + beta[tid + 256]);
    orow[tid + 512] = f2bf((v2 - mu) * rs * g[tid + 512] + beta[tid + 512]);
}

// ---------------------------------------------------------------------------
// Tiled bf16 GEMM: out[M,N] = A[M,K] @ WT[N,K]^T + bias (+ epilogue).
// 256 thr = 8 waves; block tile 128x128, k-step 32, double-buffered async
// global->LDS staging. Wave (wm 0..3, wn 0..1) owns 32x64 = 2x4 WMMA accs.
// ---------------------------------------------------------------------------
template<int OUT_BF16, int RELU6, int RESID>
__global__ __launch_bounds__(256)
void gemm_bf16_kernel(const bf16_t* __restrict__ A, const bf16_t* __restrict__ WT,
                      const float* __restrict__ bias, const float* __restrict__ resid,
                      void* __restrict__ outv, int M, int K, int N) {
    __shared__ __align__(16) bf16_t As[2][128][40];   // [buf][m][k], padded
    __shared__ __align__(16) bf16_t Bs[2][128][40];   // [buf][n][k], padded

    const int tid  = threadIdx.x;
    const int lane = tid & 31;
    const int wave = tid >> 5;
    const int half = lane >> 4;
    const int ln   = lane & 15;
    const int wm   = wave & 3;
    const int wn   = wave >> 2;
    const int mbase = blockIdx.x * 128;
    const int nbase = blockIdx.y * 128;

    v8f acc[2][4];
    #pragma unroll
    for (int i = 0; i < 2; i++)
        #pragma unroll
        for (int j = 0; j < 4; j++)
            #pragma unroll
            for (int e = 0; e < 8; e++) acc[i][j][e] = 0.0f;

    // staging map: 128 rows x 32 k, 16 bf16 per thread, identical for A and B
    const int srow = tid >> 1, sseg = (tid & 1) * 16;
    const int grow = mbase + srow;
    const bf16_t* aptr = A  + (size_t)grow * K + sseg;            // only read if grow<M
    const bf16_t* bptr = WT + (size_t)(nbase + srow) * K + sseg;  // always valid

    auto stage = [&](int k0, int buf) {
        if (grow < M) {
            async_cp16(aptr + k0,     &As[buf][srow][sseg]);
            async_cp16(aptr + k0 + 8, &As[buf][srow][sseg + 8]);
        } else {
            v4u z = 0;
            *(v4u*)&As[buf][srow][sseg]     = z;
            *(v4u*)&As[buf][srow][sseg + 8] = z;
        }
        async_cp16(bptr + k0,     &Bs[buf][srow][sseg]);
        async_cp16(bptr + k0 + 8, &Bs[buf][srow][sseg + 8]);
    };

    const int T = K >> 5;
    stage(0, 0);
    wait_async0();
    __syncthreads();

    for (int t = 0; t < T; ++t) {
        const int cur = t & 1;
        if (t + 1 < T) stage((t + 1) << 5, cur ^ 1);   // overlap with compute

        Pack16 af[2], bfr[4];
        #pragma unroll
        for (int mt = 0; mt < 2; mt++) {
            const int r = wm * 32 + mt * 16 + ln;
            af[mt].q[0] = *(const v4u*)&As[cur][r][half * 8];
            af[mt].q[1] = *(const v4u*)&As[cur][r][16 + half * 8];
        }
        #pragma unroll
        for (int nt = 0; nt < 4; nt++) {
            const int c = wn * 64 + nt * 16 + ln;
            bfr[nt].q[0] = *(const v4u*)&Bs[cur][c][half * 8];
            bfr[nt].q[1] = *(const v4u*)&Bs[cur][c][16 + half * 8];
        }
        #pragma unroll
        for (int mt = 0; mt < 2; mt++)
            #pragma unroll
            for (int nt = 0; nt < 4; nt++)
                acc[mt][nt] = wmma_bf16(af[mt].v, bfr[nt].v, acc[mt][nt]);

        wait_async0();       // next tile landed (compiler only tracks DScnt)
        __syncthreads();
    }

    // ---- epilogue: bias (+relu6) (+residual), fp32 or bf16 out ----
    #pragma unroll
    for (int mt = 0; mt < 2; mt++) {
        const int rbase = mbase + wm * 32 + mt * 16 + 8 * half;
        #pragma unroll
        for (int nt = 0; nt < 4; nt++) {
            const int col = nbase + wn * 64 + nt * 16 + ln;
            const float bv = bias[col];
            #pragma unroll
            for (int r = 0; r < 8; r++) {
                const int row = rbase + r;
                if (row < M) {
                    float v = acc[mt][nt][r] + bv;
                    if (RELU6) v = fminf(fmaxf(v, 0.0f), 6.0f);
                    if (RESID) v += resid[(size_t)row * N + col];
                    if (OUT_BF16) ((bf16_t*)outv)[(size_t)row * N + col] = f2bf(v);
                    else          ((float*)outv)[(size_t)row * N + col]  = v;
                }
            }
        }
    }
}

// ---------------------------------------------------------------------------
// Flash attention: qkv bf16 [M, 2304] (q|k|v, head-contiguous). 128 thr =
// 4 waves, each wave owns 16 q rows; online softmax over 64-key tiles.
// ---------------------------------------------------------------------------
__global__ __launch_bounds__(128)
void flash_attn_kernel(const bf16_t* __restrict__ qkv, bf16_t* __restrict__ o) {
    constexpr int NT = 577, D = 768, QK = 2304, H = 12;
    __shared__ __align__(16) bf16_t Kt[64][72];      // [key][dh]
    __shared__ __align__(16) bf16_t Vt[64][72];      // [dh][key] (transposed)
    __shared__ __align__(16) bf16_t Ps[4][16][72];   // per-wave P staging

    const int tid = threadIdx.x, lane = tid & 31, wave = tid >> 5;
    const int half = lane >> 4, ln = lane & 15;
    const int qblk = blockIdx.x;
    const int b = blockIdx.y / H, h = blockIdx.y % H;
    const size_t bbase = (size_t)b * NT;

    // Q fragments pinned in registers (A layout, Dh=64 -> 2 k-chunks)
    Pack16 qa[2];
    {
        int qr = qblk * 64 + wave * 16 + ln;
        if (qr >= NT) qr = NT - 1;
        const bf16_t* qp = qkv + (bbase + qr) * QK + h * 64;
        qa[0].q[0] = *(const v4u*)(qp + half * 8);
        qa[0].q[1] = *(const v4u*)(qp + 16 + half * 8);
        qa[1].q[0] = *(const v4u*)(qp + 32 + half * 8);
        qa[1].q[1] = *(const v4u*)(qp + 48 + half * 8);
    }

    v8f oacc[4];
    #pragma unroll
    for (int d = 0; d < 4; d++)
        #pragma unroll
        for (int e = 0; e < 8; e++) oacc[d][e] = 0.0f;
    float run_m[8], run_l[8];
    #pragma unroll
    for (int r = 0; r < 8; r++) { run_m[r] = -1e30f; run_l[r] = 0.0f; }

    const int key = tid >> 1, seg = (tid & 1) * 32;

    for (int kt0 = 0; kt0 < NT; kt0 += 64) {
        __syncthreads();
        // ---- stage K (async, layout-preserving) and V (manual transpose) ----
        const int gk = kt0 + key;
        if (gk < NT) {
            const bf16_t* kp = qkv + (bbase + gk) * QK + D + h * 64 + seg;
            const bf16_t* vp = kp + D;
            async_cp16(kp,      &Kt[key][seg]);
            async_cp16(kp + 8,  &Kt[key][seg + 8]);
            async_cp16(kp + 16, &Kt[key][seg + 16]);
            async_cp16(kp + 24, &Kt[key][seg + 24]);
            Pack16 vl0, vl1;
            vl0.q[0] = *(const v4u*)(vp);      vl0.q[1] = *(const v4u*)(vp + 8);
            vl1.q[0] = *(const v4u*)(vp + 16); vl1.q[1] = *(const v4u*)(vp + 24);
            #pragma unroll
            for (int j = 0; j < 16; j++) {
                Vt[seg + j][key]      = vl0.e[j];
                Vt[seg + 16 + j][key] = vl1.e[j];
            }
        } else {
            v4u z = 0;
            *(v4u*)&Kt[key][seg]      = z;
            *(v4u*)&Kt[key][seg + 8]  = z;
            *(v4u*)&Kt[key][seg + 16] = z;
            *(v4u*)&Kt[key][seg + 24] = z;
            const bf16_t zb = f2bf(0.0f);
            #pragma unroll
            for (int j = 0; j < 32; j++) Vt[seg + j][key] = zb;
        }
        wait_async0();
        __syncthreads();

        // ---- S = Q K^T (4 key sub-tiles of 16) ----
        float sarr[4][8], parr[4][8];
        #pragma unroll
        for (int nt = 0; nt < 4; nt++) {
            Pack16 kb0, kb1;
            const int kc = nt * 16 + ln;
            kb0.q[0] = *(const v4u*)&Kt[kc][half * 8];
            kb0.q[1] = *(const v4u*)&Kt[kc][16 + half * 8];
            kb1.q[0] = *(const v4u*)&Kt[kc][32 + half * 8];
            kb1.q[1] = *(const v4u*)&Kt[kc][48 + half * 8];
            v8f s;
            #pragma unroll
            for (int e = 0; e < 8; e++) s[e] = 0.0f;
            s = wmma_bf16(qa[0].v, kb0.v, s);
            s = wmma_bf16(qa[1].v, kb1.v, s);
            const bool valid = (kt0 + nt * 16 + ln) < NT;
            #pragma unroll
            for (int r = 0; r < 8; r++)
                sarr[nt][r] = valid ? s[r] * 0.125f : -1e30f;
        }

        // ---- online softmax (row = 8*half + r lives within 16-lane half) ----
        #pragma unroll
        for (int r = 0; r < 8; r++) {
            float tm = fmaxf(fmaxf(sarr[0][r], sarr[1][r]),
                             fmaxf(sarr[2][r], sarr[3][r]));
            tm = fmaxf(tm, __shfl_xor(tm, 1));
            tm = fmaxf(tm, __shfl_xor(tm, 2));
            tm = fmaxf(tm, __shfl_xor(tm, 4));
            tm = fmaxf(tm, __shfl_xor(tm, 8));
            const float nm   = fmaxf(run_m[r], tm);
            const float corr = __expf(run_m[r] - nm);
            run_m[r] = nm;
            float ls = 0.0f;
            #pragma unroll
            for (int nt = 0; nt < 4; nt++) {
                const float pv = __expf(sarr[nt][r] - nm);
                parr[nt][r] = pv;
                ls += pv;
            }
            ls += __shfl_xor(ls, 1);
            ls += __shfl_xor(ls, 2);
            ls += __shfl_xor(ls, 4);
            ls += __shfl_xor(ls, 8);
            run_l[r] = run_l[r] * corr + ls;
            #pragma unroll
            for (int d = 0; d < 4; d++) oacc[d][r] *= corr;
        }

        // ---- P: accumulator layout -> A-fragment layout via LDS staging ----
        #pragma unroll
        for (int nt = 0; nt < 4; nt++)
            #pragma unroll
            for (int r = 0; r < 8; r++)
                Ps[wave][8 * half + r][nt * 16 + ln] = f2bf(parr[nt][r]);
        asm volatile("s_wait_dscnt 0x0" ::: "memory");

        Pack16 pa0, pa1;
        pa0.q[0] = *(const v4u*)&Ps[wave][ln][half * 8];
        pa0.q[1] = *(const v4u*)&Ps[wave][ln][16 + half * 8];
        pa1.q[0] = *(const v4u*)&Ps[wave][ln][32 + half * 8];
        pa1.q[1] = *(const v4u*)&Ps[wave][ln][48 + half * 8];

        // ---- O += P V ----
        #pragma unroll
        for (int dt = 0; dt < 4; dt++) {
            Pack16 vb0, vb1;
            const int dc = dt * 16 + ln;
            vb0.q[0] = *(const v4u*)&Vt[dc][half * 8];
            vb0.q[1] = *(const v4u*)&Vt[dc][16 + half * 8];
            vb1.q[0] = *(const v4u*)&Vt[dc][32 + half * 8];
            vb1.q[1] = *(const v4u*)&Vt[dc][48 + half * 8];
            oacc[dt] = wmma_bf16(pa0.v, vb0.v, oacc[dt]);
            oacc[dt] = wmma_bf16(pa1.v, vb1.v, oacc[dt]);
        }
    }

    // ---- normalize and store O ----
    #pragma unroll
    for (int r = 0; r < 8; r++) {
        const int qr = qblk * 64 + wave * 16 + 8 * half + r;
        if (qr < NT) {
            const float inv = 1.0f / run_l[r];
            bf16_t* op = o + (bbase + qr) * D + h * 64;
            #pragma unroll
            for (int dt = 0; dt < 4; dt++)
                op[dt * 16 + ln] = f2bf(oacc[dt][r] * inv);
        }
    }
}

// ---------------------------------------------------------------------------
extern "C" void kernel_launch(void* const* d_in, const int* in_sizes, int n_in,
                              void* d_out, int out_size, void* d_ws, size_t ws_size,
                              hipStream_t stream) {
    (void)in_sizes; (void)n_in; (void)out_size; (void)ws_size;
    const float* x     = (const float*)d_in[0];
    const float* ln1g  = (const float*)d_in[1];
    const float* ln1b  = (const float*)d_in[2];
    const float* ln2g  = (const float*)d_in[3];
    const float* ln2b  = (const float*)d_in[4];
    const float* wqkv  = (const float*)d_in[5];
    const float* bqkv  = (const float*)d_in[6];
    const float* wproj = (const float*)d_in[7];
    const float* bproj = (const float*)d_in[8];
    const float* wfc1  = (const float*)d_in[9];
    const float* bfc1  = (const float*)d_in[10];
    const float* wfc2  = (const float*)d_in[11];
    const float* bfc2  = (const float*)d_in[12];
    float* out = (float*)d_out;

    constexpr int B = 32, N = 577, D = 768, HID = 3072, QKN = 2304, H = 12;
    constexpr int M = B * N;   // 18464 token rows

    char* p = (char*)d_ws;
    auto take = [&](size_t nbytes) {
        void* r = (void*)p;
        p += (nbytes + 255) & ~(size_t)255;
        return r;
    };
    bf16_t* wqkv_t  = (bf16_t*)take((size_t)QKN * D * 2);   // [2304][768]
    bf16_t* wproj_t = (bf16_t*)take((size_t)D * D * 2);     // [768][768]
    bf16_t* wfc1_t  = (bf16_t*)take((size_t)HID * D * 2);   // [3072][768]
    bf16_t* wfc2_t  = (bf16_t*)take((size_t)D * HID * 2);   // [768][3072]
    bf16_t* hb      = (bf16_t*)take((size_t)M * D * 2);
    bf16_t* qkv_b   = (bf16_t*)take((size_t)M * QKN * 2);
    bf16_t* o_b     = (bf16_t*)take((size_t)M * D * 2);
    bf16_t* h2b     = (bf16_t*)take((size_t)M * D * 2);
    bf16_t* fc1_b   = (bf16_t*)take((size_t)M * HID * 2);

    // weights -> bf16, transposed to [N][K] (deterministic, every launch)
    cvt_transpose_kernel<<<dim3(D/32,   QKN/32), 256, 0, stream>>>(wqkv,  wqkv_t,  D, QKN);
    cvt_transpose_kernel<<<dim3(D/32,   D/32),   256, 0, stream>>>(wproj, wproj_t, D, D);
    cvt_transpose_kernel<<<dim3(D/32,   HID/32), 256, 0, stream>>>(wfc1,  wfc1_t,  D, HID);
    cvt_transpose_kernel<<<dim3(HID/32, D/32),   256, 0, stream>>>(wfc2,  wfc2_t,  HID, D);

    const int mblk = (M + 127) / 128;

    // h = LN1(x); qkv = h @ w_qkv + b_qkv
    layernorm_kernel<<<M, 256, 0, stream>>>(x, ln1g, ln1b, hb);
    gemm_bf16_kernel<1,0,0><<<dim3(mblk, QKN/128), 256, 0, stream>>>(
        hb, wqkv_t, bqkv, nullptr, qkv_b, M, D, QKN);

    // o = softmax(q k^T / 8) v   (flash, per (b,h,64-q-row block))
    flash_attn_kernel<<<dim3((N + 63) / 64, B * H), 128, 0, stream>>>(qkv_b, o_b);

    // x1 = x + o @ w_proj + b_proj   -> d_out
    gemm_bf16_kernel<0,0,1><<<dim3(mblk, D/128), 256, 0, stream>>>(
        o_b, wproj_t, bproj, x, out, M, D, D);

    // h2 = LN2(x1); f = relu6(h2 @ w_fc1 + b_fc1)
    layernorm_kernel<<<M, 256, 0, stream>>>(out, ln2g, ln2b, h2b);
    gemm_bf16_kernel<1,1,0><<<dim3(mblk, HID/128), 256, 0, stream>>>(
        h2b, wfc1_t, bfc1, nullptr, fc1_b, M, D, HID);

    // out = x1 + f @ w_fc2 + b_fc2  (reads+writes d_out, one owner per element)
    gemm_bf16_kernel<0,0,1><<<dim3(mblk, D/128), 256, 0, stream>>>(
        fc1_b, wfc2_t, bfc2, out, out, M, HID, D);
}
